// VectorQuantization_12395275616491
// MI455X (gfx1250) — compile-verified, matrix operand order checked by reading, not run on hardware
//
#include <hip/hip_runtime.h>

typedef __attribute__((ext_vector_type(16))) __bf16 v16bf;
typedef __attribute__((ext_vector_type(8)))  float  v8f;

#define DECAY   0.99f
#define EPSV    1e-05f
#define D_DIM   512
#define N_CODES 8192
#define BL      16384          // 8 * 2048
#define M_TILE  64             // rows per block (4 waves x 16)
#define LDS_STRIDE 520         // 512 + 8 halves pad -> conflict-free A-frag reads

// ---------------- workspace layout (bytes) ----------------
#define WS_XB      0                                  // BL*D bf16      = 16 MB
#define WS_EBT     (WS_XB   + (size_t)BL*D_DIM*2)     // N*D bf16       = 8 MB
#define WS_EMBT    (WS_EBT  + (size_t)N_CODES*D_DIM*2)// N*D f32        = 16 MB
#define WS_ESQ     (WS_EMBT + (size_t)N_CODES*D_DIM*4)// N f32
#define WS_IND     (WS_ESQ  + (size_t)N_CODES*4)      // BL int
#define WS_ES      (WS_IND  + (size_t)BL*4)           // N*D f32 scatter accum
#define WS_CNT     (WS_ES   + (size_t)N_CODES*D_DIM*4)// N f32
#define WS_NTOT    (WS_CNT  + (size_t)N_CODES*4)      // 1 f32 (padded)

// ---------------- d_out layout (float elements) ----------------
#define OUT_Q      0                                   // 8388608
#define OUT_DIFF   8388608                             // 1
#define OUT_IND    8388609                             // 16384
#define OUT_NEMB   8404993                             // 4194304
#define OUT_NCS    12599297                            // 8192
#define OUT_NAVG   12607489                            // 4194304

// =============== prep kernels ===============

__global__ __launch_bounds__(256) void zero_kernel(float* __restrict__ p, int n,
                                                   float* __restrict__ diff) {
  int i = blockIdx.x * 256 + threadIdx.x;
  if (i < n) p[i] = 0.0f;
  if (i == 0) *diff = 0.0f;
}

__global__ __launch_bounds__(256) void cvt_x_kernel(const float* __restrict__ x,
                                                    __bf16* __restrict__ xb) {
  int i = (blockIdx.x * 256 + threadIdx.x) * 8;
#pragma unroll
  for (int j = 0; j < 8; ++j) xb[i + j] = (__bf16)x[i + j];
}

// embed (D,N) f32 -> embedT (N,D) f32 + ebT (N,D) bf16, LDS 32x32 tile transpose
__global__ __launch_bounds__(256) void transpose_embed_kernel(const float* __restrict__ embed,
                                                              float* __restrict__ embedT,
                                                              __bf16* __restrict__ ebT) {
  __shared__ float tile[32][33];
  int nb = blockIdx.x * 32;
  int db = blockIdx.y * 32;
  int tx = threadIdx.x & 31;
  int ty = threadIdx.x >> 5;  // 0..7
#pragma unroll
  for (int j = 0; j < 32; j += 8)
    tile[ty + j][tx] = embed[(size_t)(db + ty + j) * N_CODES + nb + tx];
  __syncthreads();
#pragma unroll
  for (int j = 0; j < 32; j += 8) {
    float v = tile[tx][ty + j];
    size_t o = (size_t)(nb + ty + j) * D_DIM + db + tx;
    embedT[o] = v;
    ebT[o]    = (__bf16)v;
  }
}

// esq[n] = sum_d embed[d][n]^2   (reads contiguous embedT rows)
__global__ __launch_bounds__(256) void esq_kernel(const float* __restrict__ embedT,
                                                  float* __restrict__ esq) {
  __shared__ float red[256];
  int n = blockIdx.x;
  float a = embedT[(size_t)n * D_DIM + threadIdx.x];
  float b = embedT[(size_t)n * D_DIM + threadIdx.x + 256];
  red[threadIdx.x] = a * a + b * b;
  __syncthreads();
  for (int s = 128; s > 0; s >>= 1) {
    if ((int)threadIdx.x < s) red[threadIdx.x] += red[threadIdx.x + s];
    __syncthreads();
  }
  if (threadIdx.x == 0) esq[n] = red[0];
}

// =============== main WMMA distance + argmin kernel ===============
// block = 128 threads (4 wave32), each wave owns 16 token rows; grid = BL/64.
// dist(m,n) = esq[n] - 2 * <x_m, e_n>  ;  ||x||^2 dropped (constant per row)
__global__ __launch_bounds__(128) void vq_argmin_kernel(const __bf16* __restrict__ xb,
                                                        const __bf16* __restrict__ ebT,
                                                        const float*  __restrict__ esq_g,
                                                        int*   __restrict__ ind_out,
                                                        float* __restrict__ indf_out) {
  extern __shared__ char smem[];
  __bf16* Ald = (__bf16*)smem;                                       // 64 x 520 bf16
  float*  esq = (float*)(smem + (size_t)M_TILE * LDS_STRIDE * 2);    // 8192 f32

  const int lane    = threadIdx.x & 31;
  const int wave    = threadIdx.x >> 5;       // 0..3
  const int hgrp    = lane >> 4;              // 0: K 0..7/16..23, rows 0..7 ; 1: +8
  const int colres  = lane & 15;
  const int rowbase = blockIdx.x * M_TILE;
  const int wrow    = wave * 16;

  // stage A tile: 64 rows x 512 bf16, 16B chunks
  for (int i = threadIdx.x; i < M_TILE * (D_DIM / 8); i += 128) {
    int r = i >> 6;                 // 64 chunks per row
    int c = i & 63;
    uint4 v = *(const uint4*)(xb + (size_t)(rowbase + r) * D_DIM + c * 8);
    *(uint4*)(Ald + (size_t)r * LDS_STRIDE + c * 8) = v;
  }
  // stage esq (8192 f32)
  for (int i = threadIdx.x; i < N_CODES / 4; i += 128)
    *(float4*)(esq + i * 4) = *(const float4*)(esq_g + i * 4);
  __syncthreads();

  float bestv[8];
  int   bestn[8];
#pragma unroll
  for (int r = 0; r < 8; ++r) { bestv[r] = 3.4e38f; bestn[r] = 0; }

  for (int nbb = 0; nbb < N_CODES; nbb += 64) {
    if (nbb + 64 < N_CODES)   // warm next code chunk (sits in L2; pull toward WGP)
      __builtin_prefetch((const void*)(ebT + (size_t)(nbb + 64 + colres) * D_DIM), 0, 0);

    v8f acc[4];
#pragma unroll
    for (int t = 0; t < 4; ++t)
#pragma unroll
      for (int r = 0; r < 8; ++r) acc[t][r] = 0.0f;

    for (int kb = 0; kb < D_DIM; kb += 32) {
      const int koff = kb + hgrp * 8;
      // A fragment: row = wrow + colres, two contiguous 16B chunks from LDS
      union { v16bf v; uint4 q[2]; } af;
      const __bf16* ap = Ald + (size_t)(wrow + colres) * LDS_STRIDE + koff;
      af.q[0] = *(const uint4*)(ap);
      af.q[1] = *(const uint4*)(ap + 16);
#pragma unroll
      for (int t = 0; t < 4; ++t) {
        const int ncol = nbb + t * 16 + colres;
        union { v16bf v; uint4 q[2]; } bfm;
        const __bf16* bp = ebT + (size_t)ncol * D_DIM + koff;
        bfm.q[0] = *(const uint4*)(bp);
        bfm.q[1] = *(const uint4*)(bp + 16);
        acc[t] = __builtin_amdgcn_wmma_f32_16x16x32_bf16(
            false, af.v, false, bfm.v, (short)0, acc[t], false, false);
      }
    }
    // fold into running argmin: lane holds col (nbb+t*16+colres), rows hgrp*8+r
#pragma unroll
    for (int t = 0; t < 4; ++t) {
      const int ncol = nbb + t * 16 + colres;
      const float e2 = esq[ncol];
#pragma unroll
      for (int r = 0; r < 8; ++r) {
        float dist = e2 - 2.0f * acc[t][r];
        if (dist < bestv[r]) { bestv[r] = dist; bestn[r] = ncol; }
      }
    }
  }

  // cross-lane argmin within each 16-lane group (rows identical inside a group)
#pragma unroll
  for (int off = 1; off < 16; off <<= 1) {
#pragma unroll
    for (int r = 0; r < 8; ++r) {
      float ov = __shfl_xor(bestv[r], off, 32);
      int   oi = __shfl_xor(bestn[r], off, 32);
      if (ov < bestv[r] || (ov == bestv[r] && oi < bestn[r])) {
        bestv[r] = ov; bestn[r] = oi;
      }
    }
  }
  if (colres == 0) {
    const int mbase = rowbase + wrow + hgrp * 8;
#pragma unroll
    for (int r = 0; r < 8; ++r) {
      ind_out[mbase + r]  = bestn[r];
      indf_out[mbase + r] = (float)bestn[r];
    }
  }
}

// =============== gather / EMA scatter ===============

// one block per token: quantize row, diff partial, scatter x into es[(n,d)], count
__global__ __launch_bounds__(256) void quantize_kernel(const float* __restrict__ x,
                                                       const float* __restrict__ embedT,
                                                       const int*   __restrict__ ind,
                                                       float* __restrict__ q_out,
                                                       float* __restrict__ diff_out,
                                                       float* __restrict__ es_ws,
                                                       float* __restrict__ cnt_ws) {
  __shared__ float red[256];
  const int m   = blockIdx.x;
  const int idx = ind[m];
  float local = 0.0f;
#pragma unroll
  for (int j = threadIdx.x; j < D_DIM; j += 256) {
    float q  = embedT[(size_t)idx * D_DIM + j];
    float xv = x[(size_t)m * D_DIM + j];
    q_out[(size_t)m * D_DIM + j] = q;       // quantize_st == quantize numerically
    float d = q - xv;
    local += d * d;
    atomicAdd(&es_ws[(size_t)idx * D_DIM + j], xv);   // coalesced row scatter
  }
  red[threadIdx.x] = local;
  __syncthreads();
  for (int s = 128; s > 0; s >>= 1) {
    if ((int)threadIdx.x < s) red[threadIdx.x] += red[threadIdx.x + s];
    __syncthreads();
  }
  if (threadIdx.x == 0) {
    atomicAdd(diff_out, red[0] * (1.0f / (float)((size_t)BL * D_DIM)));
    atomicAdd(&cnt_ws[idx], 1.0f);
  }
}

// new_cluster_size + global sum n
__global__ __launch_bounds__(256) void cluster_kernel(const float* __restrict__ cluster_size,
                                                      const float* __restrict__ cnt,
                                                      float* __restrict__ ncs_out,
                                                      float* __restrict__ ntot) {
  __shared__ float red[256];
  int n = blockIdx.x * 256 + threadIdx.x;
  float v = cluster_size[n] * DECAY + (1.0f - DECAY) * cnt[n];
  ncs_out[n] = v;
  red[threadIdx.x] = v;
  __syncthreads();
  for (int s = 128; s > 0; s >>= 1) {
    if ((int)threadIdx.x < s) red[threadIdx.x] += red[threadIdx.x + s];
    __syncthreads();
  }
  if (threadIdx.x == 0) atomicAdd(ntot, red[0]);
}

// new_embed_avg (D,N) and new_embed (D,N)
__global__ __launch_bounds__(256) void finalize_kernel(const float* __restrict__ embed_avg,
                                                       const float* __restrict__ es_ws,
                                                       const float* __restrict__ ncs,
                                                       const float* __restrict__ ntot_p,
                                                       float* __restrict__ new_embed,
                                                       float* __restrict__ new_avg) {
  int id = blockIdx.x * 256 + threadIdx.x;       // < D*N
  int n = id & (N_CODES - 1);
  int d = id >> 13;
  float avg = embed_avg[id] * DECAY + (1.0f - DECAY) * es_ws[(size_t)n * D_DIM + d];
  new_avg[id] = avg;
  float nt = *ntot_p;
  float cs = (ncs[n] + EPSV) / (nt + (float)N_CODES * EPSV) * nt;
  new_embed[id] = avg / cs;
}

// =============== launcher ===============

extern "C" void kernel_launch(void* const* d_in, const int* in_sizes, int n_in,
                              void* d_out, int out_size, void* d_ws, size_t ws_size,
                              hipStream_t stream) {
  const float* x            = (const float*)d_in[0];
  const float* embed        = (const float*)d_in[1];
  const float* cluster_size = (const float*)d_in[2];
  const float* embed_avg    = (const float*)d_in[3];
  float* out = (float*)d_out;
  char*  ws  = (char*)d_ws;

  __bf16* xb     = (__bf16*)(ws + WS_XB);
  __bf16* ebT    = (__bf16*)(ws + WS_EBT);
  float*  embedT = (float*) (ws + WS_EMBT);
  float*  esq    = (float*) (ws + WS_ESQ);
  int*    ind    = (int*)   (ws + WS_IND);
  float*  es_ws  = (float*) (ws + WS_ES);
  float*  cnt    = (float*) (ws + WS_CNT);
  float*  ntot   = (float*) (ws + WS_NTOT);

  // 1) zero scatter accumulators (es | cnt | ntot are contiguous) + diff slot
  {
    int nz = N_CODES * D_DIM + N_CODES + 4;
    zero_kernel<<<(nz + 255) / 256, 256, 0, stream>>>(es_ws, nz, out + OUT_DIFF);
  }
  // 2) x -> bf16
  cvt_x_kernel<<<(BL * D_DIM / 8) / 256, 256, 0, stream>>>(x, xb);
  // 3) embed transpose: (D,N)f32 -> (N,D) f32 + bf16
  {
    dim3 g(N_CODES / 32, D_DIM / 32);
    transpose_embed_kernel<<<g, 256, 0, stream>>>(embed, embedT, ebT);
  }
  // 4) ||e||^2 per code
  esq_kernel<<<N_CODES, 256, 0, stream>>>(embedT, esq);
  // 5) fused bf16-WMMA distance GEMM + argmin
  {
    size_t shmem = (size_t)M_TILE * LDS_STRIDE * 2 + (size_t)N_CODES * 4;  // ~99 KB
    vq_argmin_kernel<<<BL / M_TILE, 128, shmem, stream>>>(xb, ebT, esq, ind,
                                                          out + OUT_IND);
  }
  // 6) gather quantize rows, diff, scatter EMA sums
  quantize_kernel<<<BL, 256, 0, stream>>>(x, embedT, ind, out + OUT_Q,
                                          out + OUT_DIFF, es_ws, cnt);
  // 7) new_cluster_size + total
  cluster_kernel<<<N_CODES / 256, 256, 0, stream>>>(cluster_size, cnt,
                                                    out + OUT_NCS, ntot);
  // 8) new_embed_avg, new_embed
  finalize_kernel<<<(D_DIM * N_CODES) / 256, 256, 0, stream>>>(
      embed_avg, es_ws, out + OUT_NCS, ntot, out + OUT_NEMB, out + OUT_NAVG);
}
